// HierarchicalAttentionPooling_3307124818499
// MI455X (gfx1250) — matrix operand order repaired and checked by reading, not compiled, not run on hardware
//
#include <hip/hip_runtime.h>
#include <hip/hip_fp16.h>
#include <math.h>

typedef __attribute__((ext_vector_type(16))) _Float16 v16h;
typedef __attribute__((ext_vector_type(8)))  float    v8f;

#define BDIM 256
#define DD   128   // feature dim
#define HH   128   // gate hidden dim
#define NB   256   // batch
#define NN   64    // nodes
#define NT   60    // timesteps

// ---------------------------------------------------------------------------
// Pack weight matrix W1 [K=128][N=128] (fp32 row-major) into f16 WMMA
// B-fragment order for v_wmma_f32_16x16x32_f16:
//   fragment f = kt*8 + nt   (kt: K-tile of 32, nt: N-tile of 16)
//   per fragment: 32 lanes x 16 halves contiguous.
//   lanes 0-15:  N = lane,    half i -> K = kt*32 + (i&7) + (i>=8 ? 16 : 0)
//   lanes 16-31: N = lane-16, half i -> K = kt*32 + 8 + (i&7) + (i>=8 ? 16 : 0)
// ---------------------------------------------------------------------------
__global__ void pack_w1(const float* __restrict__ w_sg,
                        const float* __restrict__ w_tg,
                        __half* __restrict__ out_sg,
                        __half* __restrict__ out_tg) {
  int tid = blockIdx.x * blockDim.x + threadIdx.x;
  if (tid >= 2 * 16384) return;
  const float* src = (tid < 16384) ? w_sg : w_tg;
  __half*      dst = (tid < 16384) ? out_sg : out_tg;
  int t    = tid & 16383;
  int i    = t & 15;
  int lane = (t >> 4) & 31;
  int f    = t >> 9;            // 0..31
  int nt   = f & 7;
  int kt   = f >> 3;
  int n    = nt * 16 + (lane & 15);
  int k    = kt * 32 + (i & 7) + ((i >= 8) ? 16 : 0) + ((lane < 16) ? 0 : 8);
  dst[t] = __float2half(src[k * HH + n]);
}

// ---------------------------------------------------------------------------
// Shared attention core: given 64 fp32 rows in LDS (padded rows zeroed),
// compute gate = tanh(X*W1 + b1) . w2, softmax over first nvalid rows,
// pooled = sum_n alpha_n * x_n  ->  sred[0][d] + sred[1][d].
// ---------------------------------------------------------------------------
__device__ __forceinline__ void attn_core(
    float* sxf32,            // [64*128] fp32 rows
    __half* sxh,             // [16 frags * 32 * 16] A-fragments
    const __half* __restrict__ wfrag,   // global, B-fragment-ordered weights
    const float* __restrict__ b1g,
    const float* __restrict__ w2g,
    float* sb1, float* sw2,  // [128] each
    float* sgate,            // [64]
    float* salpha,           // [64]
    float* sred,             // [2*128]
    int nvalid)
{
  const int tid  = threadIdx.x;
  const int lane = tid & 31;
  const int w    = tid >> 5;

  if (tid < HH) { sb1[tid] = b1g[tid]; sw2[tid] = w2g[tid]; }
  if (tid < 64) sgate[tid] = 0.0f;

  // ---- stage A fragments: f16, ISA A-layout ----
  // lane l<16 of a fragment holds row l with K' in {0..7, 16..23},
  // lane l>=16 holds row l-16... (row l) with K' in {8..15, 24..31}.
  // Write 4 consecutive K per task (stays within one 8-aligned region).
  for (int task = tid; task < 64 * 32; task += BDIM) {
    int row = task >> 5;
    int kc  = task & 31;
    int k   = kc * 4;
    float4 v = *(const float4*)(sxf32 + row * DD + k);
    int mt  = row >> 4;
    int kt  = k >> 5;
    int k32 = k & 31;
    int hi  = (k32 >> 3) & 1;
    int ln  = (row & 15) + hi * 16;
    int i0  = (k32 & 7) + ((k32 >= 16) ? 8 : 0);
    union { ushort4 u; __half h[4]; } pk;
    pk.h[0] = __float2half(v.x); pk.h[1] = __float2half(v.y);
    pk.h[2] = __float2half(v.z); pk.h[3] = __float2half(v.w);
    *(ushort4*)(sxh + (((mt * 4 + kt) * 32 + ln) * 16 + i0)) = pk.u;
  }
  __syncthreads();

  // ---- WMMA GEMM: M=64, N=128, K=128 ----
  const int mt = w & 3;        // M-tile
  const int ng = w >> 2;       // N half: N-tiles ng*4 .. ng*4+3
  v8f acc0 = {}, acc1 = {}, acc2 = {}, acc3 = {};
#pragma unroll
  for (int kt = 0; kt < 4; ++kt) {
    v16h a = *(const v16h*)(sxh + ((mt * 4 + kt) * 32 + lane) * 16);
    const __half* wb = wfrag + (size_t)((kt * 8 + ng * 4) * 32 + lane) * 16;
    v16h b0 = *(const v16h*)(wb);
    v16h b1f = *(const v16h*)(wb + 1 * 32 * 16);
    v16h b2f = *(const v16h*)(wb + 2 * 32 * 16);
    v16h b3f = *(const v16h*)(wb + 3 * 32 * 16);
    acc0 = __builtin_amdgcn_wmma_f32_16x16x32_f16(false, a, false, b0,  (short)0, acc0, false, false);
    acc1 = __builtin_amdgcn_wmma_f32_16x16x32_f16(false, a, false, b1f, (short)0, acc1, false, false);
    acc2 = __builtin_amdgcn_wmma_f32_16x16x32_f16(false, a, false, b2f, (short)0, acc2, false, false);
    acc3 = __builtin_amdgcn_wmma_f32_16x16x32_f16(false, a, false, b3f, (short)0, acc3, false, false);
  }

  // ---- gate = tanh(h + b1) . w2 : reduce over N via shuffles + LDS atomics --
  // C/D layout: VGPR v -> M = v (lanes 0-15) / v+8 (lanes 16-31); N = lane&15.
#pragma unroll
  for (int j = 0; j < 4; ++j) {
    v8f acc = (j == 0) ? acc0 : (j == 1) ? acc1 : (j == 2) ? acc2 : acc3;
    int col   = (ng * 4 + j) * 16 + (lane & 15);
    float b1v = sb1[col];
    float w2v = sw2[col];
#pragma unroll
    for (int v = 0; v < 8; ++v) {
      float p = tanhf(acc[v] + b1v) * w2v;
      p += __shfl_xor(p, 1, 32);
      p += __shfl_xor(p, 2, 32);
      p += __shfl_xor(p, 4, 32);
      p += __shfl_xor(p, 8, 32);
      if ((lane & 15) == 0) {
        int row = mt * 16 + v + ((lane >> 4) << 3);
        atomicAdd(&sgate[row], p);
      }
    }
  }
  __syncthreads();

  // ---- softmax over first nvalid rows (wave 0) ----
  if (w == 0) {
    float g0 = (lane < nvalid)      ? sgate[lane]      : -INFINITY;
    float g1 = (lane + 32 < nvalid) ? sgate[lane + 32] : -INFINITY;
    float m = fmaxf(g0, g1);
#pragma unroll
    for (int off = 16; off > 0; off >>= 1) m = fmaxf(m, __shfl_xor(m, off, 32));
    float e0 = (lane < nvalid)      ? expf(g0 - m) : 0.0f;
    float e1 = (lane + 32 < nvalid) ? expf(g1 - m) : 0.0f;
    float s = e0 + e1;
#pragma unroll
    for (int off = 16; off > 0; off >>= 1) s += __shfl_xor(s, off, 32);
    float inv = 1.0f / s;
    salpha[lane]      = e0 * inv;
    salpha[lane + 32] = e1 * inv;
  }
  __syncthreads();

  // ---- pooled[d] = sum_n alpha_n * x[n][d] ----
  {
    int d  = tid & 127;
    int hf = tid >> 7;
    float s = 0.0f;
#pragma unroll 4
    for (int n = hf * 32; n < hf * 32 + 32; ++n)
      s += salpha[n] * sxf32[n * DD + d];
    sred[hf * DD + d] = s;
  }
  __syncthreads();
}

// ---------------------------------------------------------------------------
// Stage 1: one block per (b, t). Spatial softmax-attention pool over 64 nodes.
// ---------------------------------------------------------------------------
__global__ __launch_bounds__(BDIM) void spatial_kernel(
    const float* __restrict__ x, const __half* __restrict__ wfrag,
    const float* __restrict__ b1, const float* __restrict__ w2,
    float* __restrict__ pooled_out)
{
  __shared__ __align__(16) float  sxf32[NN * DD];
  __shared__ __align__(16) __half sxh[16 * 32 * 16];
  __shared__ float sb1[HH], sw2[HH], sgate[64], salpha[64], sred[2 * DD];

  const int b = blockIdx.x / NT;
  const int t = blockIdx.x % NT;
  const int tid = threadIdx.x;
  const float* xb = x + ((size_t)b * NN * NT + t) * DD;  // row n at +n*NT*DD

#pragma unroll
  for (int it = 0; it < 8; ++it) {
    int idx = tid + it * BDIM;       // 0..2047
    int row = idx >> 5;
    int j   = idx & 31;
    float4 v = *(const float4*)(xb + (size_t)row * NT * DD + j * 4);
    *(float4*)(sxf32 + row * DD + j * 4) = v;
  }
  __syncthreads();

  attn_core(sxf32, sxh, wfrag, b1, w2, sb1, sw2, sgate, salpha, sred, NN);

  if (tid < DD)
    pooled_out[((size_t)b * NT + t) * DD + tid] = sred[tid] + sred[DD + tid];
}

// ---------------------------------------------------------------------------
// Stage 2: one block per b. Temporal softmax-attention pool over 60 steps
// (rows padded to 64 with zeros; pad alphas are exactly 0).
// ---------------------------------------------------------------------------
__global__ __launch_bounds__(BDIM) void temporal_kernel(
    const float* __restrict__ pooled, const __half* __restrict__ wfrag,
    const float* __restrict__ b1, const float* __restrict__ w2,
    float* __restrict__ final_out, float* __restrict__ tw_out)
{
  __shared__ __align__(16) float  sxf32[NN * DD];
  __shared__ __align__(16) __half sxh[16 * 32 * 16];
  __shared__ float sb1[HH], sw2[HH], sgate[64], salpha[64], sred[2 * DD];

  const int b = blockIdx.x;
  const int tid = threadIdx.x;
  const float* pb = pooled + (size_t)b * NT * DD;

#pragma unroll
  for (int it = 0; it < 8; ++it) {
    int idx = tid + it * BDIM;       // 0..2047 -> 64 rows
    int row = idx >> 5;
    int j   = idx & 31;
    float4 v = (row < NT) ? *(const float4*)(pb + row * DD + j * 4)
                          : make_float4(0.f, 0.f, 0.f, 0.f);
    *(float4*)(sxf32 + row * DD + j * 4) = v;
  }
  __syncthreads();

  attn_core(sxf32, sxh, wfrag, b1, w2, sb1, sw2, sgate, salpha, sred, NT);

  if (tid < NT) tw_out[b * NT + tid] = salpha[tid];
  if (tid < DD) final_out[b * DD + tid] = sred[tid] + sred[DD + tid];
}

// ---------------------------------------------------------------------------
extern "C" void kernel_launch(void* const* d_in, const int* in_sizes, int n_in,
                              void* d_out, int out_size, void* d_ws, size_t ws_size,
                              hipStream_t stream) {
  const float* x     = (const float*)d_in[0];
  const float* sg_w1 = (const float*)d_in[1];
  const float* sg_b1 = (const float*)d_in[2];
  const float* sg_w2 = (const float*)d_in[3];
  // d_in[4] = sg_b2: softmax is shift-invariant -> drops out exactly
  const float* tg_w1 = (const float*)d_in[5];
  const float* tg_b1 = (const float*)d_in[6];
  const float* tg_w2 = (const float*)d_in[7];
  // d_in[8] = tg_b2: drops out exactly

  float* out_final = (float*)d_out;                 // [B, D]
  float* out_tw    = out_final + NB * DD;           // [B, T]

  char*   ws     = (char*)d_ws;
  __half* wsg    = (__half*)ws;                     // 16384 halves = 32KB
  __half* wtg    = (__half*)(ws + 32768);           // 32KB
  float*  pooled = (float*)(ws + 65536);            // B*T*D fp32 = 7.9MB

  pack_w1<<<(2 * 16384 + 255) / 256, 256, 0, stream>>>(sg_w1, tg_w1, wsg, wtg);
  spatial_kernel<<<NB * NT, BDIM, 0, stream>>>(x, wsg, sg_b1, sg_w2, pooled);
  temporal_kernel<<<NB, BDIM, 0, stream>>>(pooled, wtg, tg_b1, tg_w2, out_final, out_tw);
}
